// UR5_latent_model_45397804319315
// MI455X (gfx1250) — compile-verified, matrix operand order checked by reading, not run on hardware
//
#include <hip/hip_runtime.h>
#include <hip/hip_bf16.h>

// ---------------------------------------------------------------------------
// UR5 latent-model attention step for MI455X (gfx1250, wave32, WMMA).
//
//  * mask admits 1 key/row for q<128, 4 keys/row for q>=128  -> attention is
//    sparse; only the Q/K projections are real FLOPs (10.7 GFLOP).
//  * LayerNorm shift-invariance kills the ensemble mean-centering.
//  * Projections run on the WMMA pipe (v_wmma_f32_16x16x32_bf16, f32 accum).
//    When workspace allows, operands are split bf16 hi+lo and we accumulate
//    Ahi*Bhi + Ahi*Blo + Alo*Bhi  ->  ~fp32 accuracy at 3x wmma issue, still
//    within the ~10us HBM roofline (200MB @ 23.3 TB/s).
//  * SPLIT is a compile-time template parameter so the GEMM K-loop is fully
//    unrolled straight-line code (no per-tile scalar branches between WMMAs).
// ---------------------------------------------------------------------------

typedef __attribute__((ext_vector_type(16))) __bf16 v16bf;
typedef __attribute__((ext_vector_type(8)))  float  v8f;

#define BS   128
#define DM   256
#define DIMX 256
#define DIMZ 128
#define LSEQ 640   // 256 + 3*128
#define KROWS 512  // xy rows 128..639 (rows 0..127 never attended to)

// ---------------- sinusoidal positional encoding table [128, 256] ----------
__global__ void pe_kernel(float* __restrict__ pe) {
  int p = blockIdx.x;          // 0..127  (position = batch index)
  int e = threadIdx.x;         // 0..255  (feature = ensemble dim)
  const float kf = -logf(10000.0f) / 256.0f;
  float div = expf((float)(e & ~1) * kf);     // uses i = e/2 pair index
  float v = (float)p * div;
  pe[p * DM + e] = (e & 1) ? cosf(v) : sinf(v);
}

// ---------------- fp32 -> bf16 hi (+ optional lo residual) -----------------
__global__ void cvt_split(const float* __restrict__ in, __hip_bfloat16* __restrict__ hi,
                          __hip_bfloat16* __restrict__ lo, int n, int split) {
  int i = blockIdx.x * 256 + threadIdx.x;
  if (i >= n) return;
  float x = in[i];
  __hip_bfloat16 h = __float2bfloat16(x);
  hi[i] = h;
  if (split) lo[i] = __float2bfloat16(x - __bfloat162float(h));
}

// ---------------- LN(src + pe) rows, stored bf16 hi (+lo) ------------------
// grid.x = 640 rows (0..127 -> Xq row, 128..639 -> Xk row), grid.y = batch.
__global__ void __launch_bounds__(256)
build_x(const float* __restrict__ state, const float* __restrict__ o0,
        const float* __restrict__ o1, const float* __restrict__ o2,
        const float* __restrict__ Qp, const float* __restrict__ pe,
        const float* __restrict__ g, const float* __restrict__ beta,
        __hip_bfloat16* __restrict__ Xq_hi, __hip_bfloat16* __restrict__ Xk_hi,
        __hip_bfloat16* __restrict__ Xq_lo, __hip_bfloat16* __restrict__ Xk_lo,
        int split) {
  const int row = blockIdx.x;      // 0..639
  const int b   = blockIdx.y;      // 0..127
  const int e   = threadIdx.x;     // 0..255
  float v = pe[b * DM + e];
  if (row < 128) {
    v += Qp[(size_t)(128 + row) * DM + e];              // query rows 128..255
  } else {
    int r = row - 128;                                   // xy row l = 128 + r
    if (r < 128) {
      v += state[((size_t)b * DM + e) * DIMX + (128 + r)];
    } else {
      int m = (r - 128) >> 7, j = (r - 128) & 127;
      const float* o = (m == 0) ? o0 : (m == 1) ? o1 : o2;
      v += o[((size_t)b * DM + e) * DIMZ + j];
    }
  }
  __shared__ float s1[256], s2[256];
  s1[e] = v; s2[e] = v * v;
  __syncthreads();
  for (int s = 128; s > 0; s >>= 1) {
    if (e < s) { s1[e] += s1[e + s]; s2[e] += s2[e + s]; }
    __syncthreads();
  }
  float mean = s1[0] * (1.0f / 256.0f);
  float var  = s2[0] * (1.0f / 256.0f) - mean * mean;
  float rs   = rsqrtf(var + 1e-5f);
  float x = (v - mean) * rs * g[e] + beta[e];
  __hip_bfloat16 h = __float2bfloat16(x);
  size_t idx = (row < 128) ? ((size_t)b * 128 + row) * DM + e
                           : ((size_t)b * KROWS + (row - 128)) * DM + e;
  if (row < 128) Xq_hi[idx] = h; else Xk_hi[idx] = h;
  if (split) {
    __hip_bfloat16 l = __float2bfloat16(x - __bfloat162float(h));
    if (row < 128) Xq_lo[idx] = l; else Xk_lo[idx] = l;
  }
}

// ---------------- bf16(x3) WMMA GEMM: C[b] = X[b] @ W^T + bias -------------
// X: [B, R, 256] bf16 row-major.  W: [256, 256] bf16 row-major (out, in).
// One wave computes a 16x64 tile; K-loop 8 x 32 with v_wmma_f32_16x16x32_bf16.
// Fragment addressing follows the CDNA5 16-bit layouts: each lane carries two
// 16-byte chunks at k0 + 8*half and k0 + 16 + 8*half of its row.
// SPLIT==1: accumulate Ahi*Bhi + Ahi*Blo + Alo*Bhi (~fp32 accuracy, 3x wmma).
template <int SPLIT>
__global__ void __launch_bounds__(32)
gemm_bf16_wmma(const __hip_bfloat16* __restrict__ Xhi, const __hip_bfloat16* __restrict__ Xlo,
               const __hip_bfloat16* __restrict__ Whi, const __hip_bfloat16* __restrict__ Wlo,
               const float* __restrict__ bias, float* __restrict__ C, int R) {
  const int lane = threadIdx.x;        // 0..31
  const int half = lane >> 4;          // 0/1
  const int l16  = lane & 15;
  const int mt = blockIdx.x;           // M tile (16 rows)
  const int ng = blockIdx.y;           // N group (64 cols)
  const int b  = blockIdx.z;

  const size_t xoff = ((size_t)b * R + mt * 16 + l16) * DM;
  const int co = half * 8;             // lane's chunk column offset
  v8f acc[4] = {v8f(0.f), v8f(0.f), v8f(0.f), v8f(0.f)};

  union ABU { v16bf v; uint4 u[2]; };
#pragma unroll
  for (int k0 = 0; k0 < DM; k0 += 32) {
    ABU ahi, alo;
    ahi.u[0] = *(const uint4*)(Xhi + xoff + k0 + co);
    ahi.u[1] = *(const uint4*)(Xhi + xoff + k0 + 16 + co);
    if (SPLIT) {
      alo.u[0] = *(const uint4*)(Xlo + xoff + k0 + co);
      alo.u[1] = *(const uint4*)(Xlo + xoff + k0 + 16 + co);
    }
#pragma unroll
    for (int t = 0; t < 4; ++t) {
      const size_t woff = (size_t)(ng * 64 + t * 16 + l16) * DM;  // B[k][n]=W[n][k]
      ABU bhi, blo;
      bhi.u[0] = *(const uint4*)(Whi + woff + k0 + co);
      bhi.u[1] = *(const uint4*)(Whi + woff + k0 + 16 + co);
      if (SPLIT) {
        blo.u[0] = *(const uint4*)(Wlo + woff + k0 + co);
        blo.u[1] = *(const uint4*)(Wlo + woff + k0 + 16 + co);
      }
      acc[t] = __builtin_amdgcn_wmma_f32_16x16x32_bf16(
          false, ahi.v, false, bhi.v, (short)0, acc[t], false, false);
      if (SPLIT) {
        acc[t] = __builtin_amdgcn_wmma_f32_16x16x32_bf16(
            false, ahi.v, false, blo.v, (short)0, acc[t], false, false);
        acc[t] = __builtin_amdgcn_wmma_f32_16x16x32_bf16(
            false, alo.v, false, bhi.v, (short)0, acc[t], false, false);
      }
    }
  }
  // epilogue: + bias, store f32.  C layout: VGPR r -> M = mt*16 + r + 8*half,
  // N = ng*64 + t*16 + (lane&15).
#pragma unroll
  for (int t = 0; t < 4; ++t) {
    const int n = ng * 64 + t * 16 + l16;
    const float bv = bias[n];
#pragma unroll
    for (int r = 0; r < 8; ++r) {
      const int m = mt * 16 + r + 8 * half;
      C[((size_t)b * R + m) * DM + n] = acc[t][r] + bv;
    }
  }
}

// ---------------- atten zero-fill + identity rows (q < 128) ----------------
__global__ void fill_atten(float* __restrict__ atten) {
  size_t idx = (size_t)blockIdx.x * 256 + threadIdx.x;   // < 128*256*640
  int l = (int)(idx % LSEQ);
  int q = (int)((idx / LSEQ) % DIMX);
  atten[idx] = (q < 128 && l == q) ? 1.0f : 0.0f;
}

// ---------------- new_state passthrough for q < 128 ------------------------
__global__ void ns_copy(const float* __restrict__ state, float* __restrict__ out) {
  int idx = blockIdx.x * 256 + threadIdx.x;  // < 128*256*128
  int q = idx & 127, e = (idx >> 7) & 255, b = idx >> 15;
  size_t o = ((size_t)b * DM + e) * DIMX + q;
  out[o] = state[o];
}

// ---------------- sparse scores -> softmax(4) -> head mean -----------------
// grid (j=0..127, b=0..127).  q = 128+j; keys are Ck rows {j,128+j,256+j,384+j}
// (xy rows q, 256+j, 384+j, 512+j).
__global__ void __launch_bounds__(256)
scores_kernel(const float* __restrict__ Cq, const float* __restrict__ Ck,
              float* __restrict__ alpha_out, float* __restrict__ atten) {
  const int j = blockIdx.x, b = blockIdx.y, e = threadIdx.x;
  __shared__ float sp[4][256];
  __shared__ float sh[4][4];     // [head][key]
  const float qv = Cq[((size_t)b * 128 + j) * DM + e];
  const float* ckb = Ck + (size_t)b * KROWS * DM;
#pragma unroll
  for (int t = 0; t < 4; ++t)
    sp[t][e] = qv * ckb[(size_t)(t * 128 + j) * DM + e];
  __syncthreads();
  if (e < 16) {                  // 4 heads x 4 keys, 64-wide dot each
    int h = e >> 2, t = e & 3;
    float s = 0.f;
    for (int d = 0; d < 64; ++d) s += sp[t][h * 64 + d];
    sh[h][t] = s * 0.125f;       // / sqrt(64)
  }
  __syncthreads();
  if (e < 4) {                   // softmax over the 4 admitted keys, head e
    float m = fmaxf(fmaxf(sh[e][0], sh[e][1]), fmaxf(sh[e][2], sh[e][3]));
    float e0 = expf(sh[e][0] - m), e1 = expf(sh[e][1] - m);
    float e2 = expf(sh[e][2] - m), e3 = expf(sh[e][3] - m);
    float d = e0 + e1 + e2 + e3;
    sh[e][0] = e0 / d; sh[e][1] = e1 / d; sh[e][2] = e2 / d; sh[e][3] = e3 / d;
  }
  __syncthreads();
  if (e < 4) {                   // head-average; scatter the 4 atten entries
    float a = 0.25f * (sh[0][e] + sh[1][e] + sh[2][e] + sh[3][e]);
    alpha_out[((size_t)b * 128 + j) * 4 + e] = a;
    atten[(size_t)(b * DIMX + 128 + j) * LSEQ + (128 + e * 128 + j)] = a;
  }
}

// ---------------- new_state[q>=128] = 4-term blend of raw columns ----------
__global__ void combine(const float* __restrict__ state, const float* __restrict__ o0,
                        const float* __restrict__ o1, const float* __restrict__ o2,
                        const float* __restrict__ alpha, float* __restrict__ out) {
  int idx = blockIdx.x * 256 + threadIdx.x;  // < 128*256*128
  int j = idx & 127, e = (idx >> 7) & 255, b = idx >> 15;
  const float* a = alpha + ((size_t)b * 128 + j) * 4;
  size_t so = (size_t)b * DM + e;
  float v = a[0] * state[so * DIMX + 128 + j]
          + a[1] * o0[so * DIMZ + j]
          + a[2] * o1[so * DIMZ + j]
          + a[3] * o2[so * DIMZ + j];
  out[so * DIMX + 128 + j] = v;
}

// ---------------------------------------------------------------------------
extern "C" void kernel_launch(void* const* d_in, const int* in_sizes, int n_in,
                              void* d_out, int out_size, void* d_ws, size_t ws_size,
                              hipStream_t stream) {
  (void)in_sizes; (void)n_in; (void)out_size;
  const float* state = (const float*)d_in[0];
  const float* obs0  = (const float*)d_in[1];
  const float* obs1  = (const float*)d_in[2];
  const float* obs2  = (const float*)d_in[3];
  const float* Qp    = (const float*)d_in[4];
  const float* ln_g  = (const float*)d_in[5];
  const float* ln_b  = (const float*)d_in[6];
  const float* Wq    = (const float*)d_in[7];
  const float* bq    = (const float*)d_in[8];
  const float* Wk    = (const float*)d_in[9];
  const float* bk    = (const float*)d_in[10];

  // workspace layout (bytes).  Baseline (hi-only) prefix: ~126.5 MB.
  // Split-precision lo-arrays appended after: total ~160.9 MB.
  char* ws = (char*)d_ws;
  float*          pe    = (float*)(ws + 0);                      // 131072
  __hip_bfloat16* Wq_hi = (__hip_bfloat16*)(ws + 131072);        // 131072
  __hip_bfloat16* Wk_hi = (__hip_bfloat16*)(ws + 262144);        // 131072
  __hip_bfloat16* Xq_hi = (__hip_bfloat16*)(ws + 393216);        // 8388608
  __hip_bfloat16* Xk_hi = (__hip_bfloat16*)(ws + 8781824);       // 33554432
  float*          Cq    = (float*)(ws + 42336256);               // 16777216
  float*          Ck    = (float*)(ws + 59113472);               // 67108864
  float*          alpha = (float*)(ws + 126222336);              // 262144
  __hip_bfloat16* Wq_lo = (__hip_bfloat16*)(ws + 126484480);     // 131072
  __hip_bfloat16* Wk_lo = (__hip_bfloat16*)(ws + 126615552);     // 131072
  __hip_bfloat16* Xq_lo = (__hip_bfloat16*)(ws + 126746624);     // 8388608
  __hip_bfloat16* Xk_lo = (__hip_bfloat16*)(ws + 135135232);     // 33554432
  const size_t NEED_SPLIT = 168689664;
  const int split = (ws_size >= NEED_SPLIT) ? 1 : 0;   // constant per run

  float* out_ns = (float*)d_out;                           // [128,256,256]
  float* out_at = out_ns + (size_t)BS * DM * DIMX;         // [128,256,640]

  pe_kernel<<<128, 256, 0, stream>>>(pe);
  cvt_split<<<256, 256, 0, stream>>>(Wq, Wq_hi, Wq_lo, DM * DM, split);
  cvt_split<<<256, 256, 0, stream>>>(Wk, Wk_hi, Wk_lo, DM * DM, split);
  build_x<<<dim3(640, 128), 256, 0, stream>>>(state, obs0, obs1, obs2, Qp, pe,
                                              ln_g, ln_b, Xq_hi, Xk_hi,
                                              Xq_lo, Xk_lo, split);
  if (split) {
    gemm_bf16_wmma<1><<<dim3(8, 4, 128), 32, 0, stream>>>(Xq_hi, Xq_lo, Wq_hi, Wq_lo,
                                                          bq, Cq, 128);
    gemm_bf16_wmma<1><<<dim3(32, 4, 128), 32, 0, stream>>>(Xk_hi, Xk_lo, Wk_hi, Wk_lo,
                                                           bk, Ck, KROWS);
  } else {
    gemm_bf16_wmma<0><<<dim3(8, 4, 128), 32, 0, stream>>>(Xq_hi, Xq_lo, Wq_hi, Wq_lo,
                                                          bq, Cq, 128);
    gemm_bf16_wmma<0><<<dim3(32, 4, 128), 32, 0, stream>>>(Xk_hi, Xk_lo, Wk_hi, Wk_lo,
                                                           bk, Ck, KROWS);
  }
  fill_atten<<<(BS * DIMX * LSEQ) / 256, 256, 0, stream>>>(out_at);
  ns_copy<<<(BS * DM * 128) / 256, 256, 0, stream>>>(state, out_ns);
  scores_kernel<<<dim3(128, 128), 256, 0, stream>>>(Cq, Ck, alpha, out_at);
  combine<<<(BS * DM * 128) / 256, 256, 0, stream>>>(state, obs0, obs1, obs2,
                                                     alpha, out_ns);
}
